// CLIP_70377334113068
// MI455X (gfx1250) — matrix operand order repaired and checked by reading, not compiled, not run on hardware
//
#include <hip/hip_runtime.h>
#include <hip/hip_bf16.h>

typedef __attribute__((ext_vector_type(16))) _Float16 v16h;
typedef __attribute__((ext_vector_type(8)))  float    v8f;

#define BZ   16
#define SEQ  197
#define SPAD 224            // SEQ padded to multiple of 32 for K of att@V
#define DIM  768
#define NHD  12
#define HDM  64
#define FFD  3072
#define NTOK (BZ * SEQ)     // 3152
#define NBH  (BZ * NHD)     // 192
#define NEXP 8
#define NOUT 512
#define NPATCH 196
#define PROWS (BZ * NPATCH) // 3136

#define CVT4(f, i, vv)                                                     \
  {                                                                        \
    f[(i) + 0] = (_Float16)(vv).x; f[(i) + 1] = (_Float16)(vv).y;          \
    f[(i) + 2] = (_Float16)(vv).z; f[(i) + 3] = (_Float16)(vv).w;          \
  }

// ---------------------------------------------------------------------------
// WMMA fragment loaders (fp32 memory -> f16 fragments). K%32==0 everywhere,
// row/col indices are pre-clamped (garbage rows/cols never stored).
// A 16x32: lanes0-15 row=lane, elems {K0..7,K16..23}; lanes16-31 {K8..15,K24..31}
// B 32x16: col=lane&15, elems {K0..7,K16..23}; lanes16-31 {K8..15,K24..31} (same as A)
// ---------------------------------------------------------------------------
__device__ __forceinline__ v16h load_a_frag(const float* __restrict__ A, long lda,
                                            long row, int kb) {
  const float* rp = A + row * lda + kb;
  float4 x0 = *(const float4*)(rp);
  float4 x1 = *(const float4*)(rp + 4);
  float4 x2 = *(const float4*)(rp + 16);
  float4 x3 = *(const float4*)(rp + 20);
  v16h f;
  CVT4(f, 0, x0); CVT4(f, 4, x1); CVT4(f, 8, x2); CVT4(f, 12, x3);
  return f;
}

__device__ __forceinline__ v16h load_b_frag(const float* __restrict__ W, long ldw,
                                            long col, int kb) {
  const float* rp = W + col * ldw + kb;
  float4 x0 = *(const float4*)(rp);
  float4 x1 = *(const float4*)(rp + 4);
  float4 x2 = *(const float4*)(rp + 16);
  float4 x3 = *(const float4*)(rp + 20);
  v16h f;
  CVT4(f, 0, x0); CVT4(f, 4, x1); CVT4(f, 8, x2); CVT4(f, 12, x3);
  return f;
}

// ---------------------------------------------------------------------------
// Generic batched GEMM with optional gather/scatter (MoE expert dispatch):
//   rows   = gatherA ? token list : identity   (A-row indirection)
//   C row  = scatterC ? scatterC[row] : row    (C-row scatter)
//   Mdyn   = dynM ? min(M, *dynM) : M          (device-side row count)
//   C = [res +] [rowscale[row] *] act(A * W^T + bias)   (+= if accum)
// W stored [N,K] row-major. Requires K % 32 == 0.
// Block 256 = 8 waves (4M x 2N); wave tile 32x32 (2x2 wmma); block tile 128x64.
// ---------------------------------------------------------------------------
__global__ __launch_bounds__(256) void gemm_kernel(
    const float* __restrict__ A, long lda, long strideA,
    const float* __restrict__ W, long ldw, long strideW,
    const float* __restrict__ bias,
    float* __restrict__ C, long ldc, long strideC,
    const float* __restrict__ residual, long ldr, long strideR,
    const float* __restrict__ rowscale,
    const int* __restrict__ gatherA, const int* __restrict__ scatterC,
    const int* __restrict__ dynM,
    int M, int N, int K, int act, int accum) {
  int Mdyn = M;
  if (dynM) { int c = *dynM; Mdyn = (c < M) ? c : M; }
  int lane = threadIdx.x & 31;
  int wave = threadIdx.x >> 5;
  int wm = wave & 3, wn = wave >> 2;
  long row0 = (long)blockIdx.y * 128 + wm * 32;
  long col0 = (long)blockIdx.x * 64 + wn * 32;
  if (row0 >= Mdyn || col0 >= N) return;  // wave-uniform (handles Mdyn==0)
  long z = blockIdx.z;
  const float* Ab = A + z * strideA;
  const float* Wb = W + z * strideW;
  float* Cb = C + z * strideC;

  long ar0 = row0 + (lane & 15);        if (ar0 > Mdyn - 1) ar0 = Mdyn - 1;
  long ar1 = row0 + 16 + (lane & 15);   if (ar1 > Mdyn - 1) ar1 = Mdyn - 1;
  if (gatherA) { ar0 = gatherA[ar0]; ar1 = gatherA[ar1]; }
  long bc0 = col0 + (lane & 15);        if (bc0 > N - 1) bc0 = N - 1;
  long bc1 = col0 + 16 + (lane & 15);   if (bc1 > N - 1) bc1 = N - 1;
  int aoff = (lane & 16) ? 8 : 0;
  int boff = (lane & 16) ? 8 : 0;

  v8f acc00 = {}, acc01 = {}, acc10 = {}, acc11 = {};
  for (int k0 = 0; k0 < K; k0 += 32) {
    v16h a0 = load_a_frag(Ab, lda, ar0, k0 + aoff);
    v16h a1 = load_a_frag(Ab, lda, ar1, k0 + aoff);
    v16h b0 = load_b_frag(Wb, ldw, bc0, k0 + boff);
    v16h b1 = load_b_frag(Wb, ldw, bc1, k0 + boff);
    acc00 = __builtin_amdgcn_wmma_f32_16x16x32_f16(false, a0, false, b0,
                                                   (short)0, acc00, false, false);
    acc01 = __builtin_amdgcn_wmma_f32_16x16x32_f16(false, a0, false, b1,
                                                   (short)0, acc01, false, false);
    acc10 = __builtin_amdgcn_wmma_f32_16x16x32_f16(false, a1, false, b0,
                                                   (short)0, acc10, false, false);
    acc11 = __builtin_amdgcn_wmma_f32_16x16x32_f16(false, a1, false, b1,
                                                   (short)0, acc11, false, false);
  }

#pragma unroll
  for (int rt = 0; rt < 2; ++rt) {
    long rb = row0 + rt * 16 + ((lane & 16) ? 8 : 0);
#pragma unroll
    for (int t = 0; t < 2; ++t) {
      long col = col0 + t * 16 + (lane & 15);
      if (col >= N) continue;
      float bv = bias ? bias[col] : 0.0f;
#pragma unroll
      for (int r = 0; r < 8; ++r) {
        long row = rb + r;
        if (row >= Mdyn) continue;
        float v = (rt ? (t ? acc11[r] : acc10[r])
                      : (t ? acc01[r] : acc00[r])) + bv;
        if (act == 1) v = v / (1.0f + __expf(-1.702f * v));  // quickGELU
        if (rowscale) v *= rowscale[row];
        long crow = scatterC ? (long)scatterC[row] : row;
        float* cp = Cb + crow * ldc + col;
        if (residual) v += residual[z * strideR + row * ldr + col];
        if (accum) v += *cp;
        *cp = v;
      }
    }
  }
}

// ---------------------------------------------------------------------------
// Patch-embed conv as im2col GEMM. x:[16,3,224,224], Wc:[768,768], h:[16,197,768]
// Exact tiles: M=3136, N=768, K=768.
// ---------------------------------------------------------------------------
__device__ __forceinline__ v16h load_patch_frag(const float* __restrict__ x,
                                                int row0, int k0, int lane) {
  int rr = row0 + (lane & 15);
  int b = rr / NPATCH, p = rr % NPATCH;
  int py = p / 14, px = p % 14;
  int kb = k0 + ((lane & 16) ? 8 : 0);
  // group of 8 contiguous k shares (c,ky): k = c*256 + ky*16 + kx, kb%16 in {0,8}
  int c0 = kb >> 8, ky0 = (kb >> 4) & 15, kx0 = kb & 15;
  const float* p0 = x + ((long)(b * 3 + c0) * 224 + py * 16 + ky0) * 224 + px * 16 + kx0;
  int k2 = kb + 16;
  int c1 = k2 >> 8, ky1 = (k2 >> 4) & 15, kx1 = k2 & 15;
  const float* p1 = x + ((long)(b * 3 + c1) * 224 + py * 16 + ky1) * 224 + px * 16 + kx1;
  float4 a0 = *(const float4*)(p0);
  float4 a1 = *(const float4*)(p0 + 4);
  float4 a2 = *(const float4*)(p1);
  float4 a3 = *(const float4*)(p1 + 4);
  v16h f;
  CVT4(f, 0, a0); CVT4(f, 4, a1); CVT4(f, 8, a2); CVT4(f, 12, a3);
  return f;
}

__global__ __launch_bounds__(256) void patch_gemm_kernel(
    const float* __restrict__ x, const float* __restrict__ Wc,
    float* __restrict__ h) {
  int lane = threadIdx.x & 31;
  int wave = threadIdx.x >> 5;
  int wm = wave & 3, wn = wave >> 2;
  int row0 = blockIdx.y * 64 + wm * 16;        // 49 blocks * 64 = 3136 exact
  int col0 = blockIdx.x * 64 + wn * 32;        // 12 blocks * 64 = 768 exact
  int boff = (lane & 16) ? 8 : 0;
  v8f acc0 = {}, acc1 = {};
  for (int k0 = 0; k0 < DIM; k0 += 32) {
    v16h a  = load_patch_frag(x, row0, k0, lane);
    v16h b0 = load_b_frag(Wc, DIM, col0 + (lane & 15), k0 + boff);
    v16h b1 = load_b_frag(Wc, DIM, col0 + 16 + (lane & 15), k0 + boff);
    acc0 = __builtin_amdgcn_wmma_f32_16x16x32_f16(false, a, false, b0,
                                                  (short)0, acc0, false, false);
    acc1 = __builtin_amdgcn_wmma_f32_16x16x32_f16(false, a, false, b1,
                                                  (short)0, acc1, false, false);
  }
  int rbase = row0 + ((lane & 16) ? 8 : 0);
#pragma unroll
  for (int t = 0; t < 2; ++t) {
    int col = col0 + t * 16 + (lane & 15);
#pragma unroll
    for (int r = 0; r < 8; ++r) {
      int row = rbase + r;
      int b_ = row / NPATCH, p = row % NPATCH;
      h[((long)(b_ * SEQ + 1 + p)) * DIM + col] = (t ? acc1[r] : acc0[r]);
    }
  }
}

// ---------------------------------------------------------------------------
// Elementwise / reduction kernels
// ---------------------------------------------------------------------------
__global__ void add_cls_pos_kernel(float* __restrict__ h,
                                   const float* __restrict__ cls,
                                   const float* __restrict__ pos) {
  long i = (long)blockIdx.x * blockDim.x + threadIdx.x;
  if (i >= (long)NTOK * DIM) return;
  int d = (int)(i % DIM);
  long t = i / DIM;
  int s = (int)(t % SEQ);
  if (s == 0) h[i] = cls[d] + pos[d];
  else        h[i] += pos[(long)s * DIM + d];
}

__global__ __launch_bounds__(256) void ln_kernel(
    const float* __restrict__ in, long ld_in,
    const float* __restrict__ w, const float* __restrict__ b,
    float* __restrict__ out, long ld_out, int D) {
  long row = blockIdx.x;
  const float* x = in + row * ld_in;
  __shared__ float red[256];
  float s = 0.0f;
  for (int i = threadIdx.x; i < D; i += 256) s += x[i];
  red[threadIdx.x] = s; __syncthreads();
  for (int off = 128; off; off >>= 1) {
    if (threadIdx.x < off) red[threadIdx.x] += red[threadIdx.x + off];
    __syncthreads();
  }
  float mean = red[0] / D; __syncthreads();
  float v = 0.0f;
  for (int i = threadIdx.x; i < D; i += 256) { float d0 = x[i] - mean; v += d0 * d0; }
  red[threadIdx.x] = v; __syncthreads();
  for (int off = 128; off; off >>= 1) {
    if (threadIdx.x < off) red[threadIdx.x] += red[threadIdx.x + off];
    __syncthreads();
  }
  float inv = rsqrtf(red[0] / D + 1e-5f);
  for (int i = threadIdx.x; i < D; i += 256)
    out[row * ld_out + i] = (x[i] - mean) * inv * w[i] + b[i];
}

// qkv [T,2304] -> q,k: [192][197][64]; v transposed+padded: vT [192][64][224]
__global__ void split_qkv_kernel(const float* __restrict__ qkv,
                                 float* __restrict__ q, float* __restrict__ k,
                                 float* __restrict__ vT) {
  long i = (long)blockIdx.x * blockDim.x + threadIdx.x;
  if (i >= (long)NTOK * 3 * DIM) return;
  long t = i / (3 * DIM);
  int c = (int)(i % (3 * DIM));
  int which = c / DIM, cc = c % DIM;
  int hh = cc / HDM, d = cc % HDM;
  int b = (int)(t / SEQ), s = (int)(t % SEQ);
  float val = qkv[i];
  long zh = (long)(b * NHD + hh);
  if (which == 0)      q[(zh * SEQ + s) * HDM + d] = val;
  else if (which == 1) k[(zh * SEQ + s) * HDM + d] = val;
  else                 vT[(zh * HDM + d) * SPAD + s] = val;
}

__global__ void zero_vpad_kernel(float* __restrict__ vT) {
  long i = (long)blockIdx.x * blockDim.x + threadIdx.x;
  if (i >= (long)NBH * HDM * (SPAD - SEQ)) return;
  long zh = i / (SPAD - SEQ);
  int s = SEQ + (int)(i % (SPAD - SEQ));
  vT[zh * SPAD + s] = 0.0f;
}

__global__ void merge_heads_kernel(const float* __restrict__ o,
                                   float* __restrict__ out) {
  long i = (long)blockIdx.x * blockDim.x + threadIdx.x;
  if (i >= (long)NTOK * DIM) return;
  long t = i / DIM;
  int c = (int)(i % DIM);
  int hh = c / HDM, d = c % HDM;
  int b = (int)(t / SEQ), s = (int)(t % SEQ);
  out[i] = o[(((long)(b * NHD + hh)) * SEQ + s) * HDM + d];
}

// proj [768][512] -> projT [512][768] so the final GEMM uses the standard path
__global__ void transpose_proj_kernel(const float* __restrict__ p,
                                      float* __restrict__ pt) {
  long i = (long)blockIdx.x * blockDim.x + threadIdx.x;
  if (i >= (long)DIM * NOUT) return;
  int n = (int)(i % NOUT);
  int k = (int)(i / NOUT);
  pt[(long)n * DIM + k] = p[i];
}

// softmax over first n entries of rows of width npad; zero-fill n..npad
__global__ __launch_bounds__(256) void softmax_rows_kernel(
    float* __restrict__ a, long nrows, int n, int npad, float scale) {
  long row = (long)blockIdx.x * 8 + (threadIdx.x >> 5);
  int lane = threadIdx.x & 31;
  if (row >= nrows) return;
  float* p = a + row * (long)npad;
  float mx = -1e30f;
  for (int i = lane; i < n; i += 32) mx = fmaxf(mx, p[i] * scale);
  for (int m = 16; m; m >>= 1) mx = fmaxf(mx, __shfl_xor(mx, m, 32));
  float sum = 0.0f;
  for (int i = lane; i < n; i += 32) {
    float e = __expf(p[i] * scale - mx);
    p[i] = e; sum += e;
  }
  for (int m = 16; m; m >>= 1) sum += __shfl_xor(sum, m, 32);
  float inv = 1.0f / sum;
  for (int i = lane; i < n; i += 32) p[i] *= inv;
  for (int i = n + lane; i < npad; i += 32) p[i] = 0.0f;
}

__global__ void zero_cnt_kernel(int* __restrict__ cnt) {
  if (threadIdx.x < NEXP && blockIdx.x == 0) cnt[threadIdx.x] = 0;
}

// top-2 softmax routing -> per-expert compacted token lists.
// cnt[e]: #tokens for expert e; idx[e][slot]: token id; wsel[e][slot]: weight.
__global__ void route_topk_kernel(const float* __restrict__ logits,
                                  int* __restrict__ cnt, int* __restrict__ idx,
                                  float* __restrict__ wsel, long T) {
  long t = (long)blockIdx.x * blockDim.x + threadIdx.x;
  if (t >= T) return;
  float l[NEXP];
  float mx = -1e30f;
#pragma unroll
  for (int e = 0; e < NEXP; ++e) { l[e] = logits[t * NEXP + e]; mx = fmaxf(mx, l[e]); }
  float sum = 0.0f;
#pragma unroll
  for (int e = 0; e < NEXP; ++e) { l[e] = __expf(l[e] - mx); sum += l[e]; }
#pragma unroll
  for (int e = 0; e < NEXP; ++e) l[e] /= sum;
  int i1 = 0;
#pragma unroll
  for (int e = 1; e < NEXP; ++e) if (l[e] > l[i1]) i1 = e;
  int i2 = (i1 == 0) ? 1 : 0;
#pragma unroll
  for (int e = 0; e < NEXP; ++e) if (e != i1 && l[e] > l[i2]) i2 = e;
  float inv = 1.0f / (l[i1] + l[i2]);
  int s1 = atomicAdd(&cnt[i1], 1);
  idx[(long)i1 * T + s1] = (int)t;
  wsel[(long)i1 * T + s1] = l[i1] * inv;
  int s2 = atomicAdd(&cnt[i2], 1);
  idx[(long)i2 * T + s2] = (int)t;
  wsel[(long)i2 * T + s2] = l[i2] * inv;
}

// ---------------------------------------------------------------------------
// Host orchestration
// ---------------------------------------------------------------------------
extern "C" void kernel_launch(void* const* d_in, const int* in_sizes, int n_in,
                              void* d_out, int out_size, void* d_ws, size_t ws_size,
                              hipStream_t stream) {
  (void)in_sizes; (void)n_in; (void)out_size; (void)ws_size;
  const float* x          = (const float*)d_in[0];
  const float* conv_w     = (const float*)d_in[1];
  const float* cls_emb    = (const float*)d_in[2];
  const float* pos_emb    = (const float*)d_in[3];
  const float* ln_pre_w   = (const float*)d_in[4];
  const float* ln_pre_b   = (const float*)d_in[5];
  const float* d_ln1_w    = (const float*)d_in[6];
  const float* d_ln1_b    = (const float*)d_in[7];
  const float* d_ain_w    = (const float*)d_in[8];
  const float* d_ain_b    = (const float*)d_in[9];
  const float* d_aout_w   = (const float*)d_in[10];
  const float* d_aout_b   = (const float*)d_in[11];
  const float* d_ln2_w    = (const float*)d_in[12];
  const float* d_ln2_b    = (const float*)d_in[13];
  const float* d_fc_w     = (const float*)d_in[14];
  const float* d_fc_b     = (const float*)d_in[15];
  const float* d_pj_w     = (const float*)d_in[16];
  const float* d_pj_b     = (const float*)d_in[17];
  const float* m_ln1_w    = (const float*)d_in[18];
  const float* m_ln1_b    = (const float*)d_in[19];
  const float* m_ain_w    = (const float*)d_in[20];
  const float* m_ain_b    = (const float*)d_in[21];
  const float* m_aout_w   = (const float*)d_in[22];
  const float* m_aout_b   = (const float*)d_in[23];
  const float* m_ln2_w    = (const float*)d_in[24];
  const float* m_ln2_b    = (const float*)d_in[25];
  const float* m_gate_w   = (const float*)d_in[26];
  const float* m_fc_w     = (const float*)d_in[27];
  const float* m_fc_b     = (const float*)d_in[28];
  const float* m_pj_w     = (const float*)d_in[29];
  const float* m_pj_b     = (const float*)d_in[30];
  const float* ln_post_w  = (const float*)d_in[31];
  const float* ln_post_b  = (const float*)d_in[32];
  const float* proj       = (const float*)d_in[33];
  float* out = (float*)d_out;

  const long TD   = (long)NTOK * DIM;            // 2,420,736
  const long TBIG = (long)NTOK * FFD;            // 9,682,944
  const long THD  = (long)NBH * SEQ * HDM;       // 2,420,736
  const long TVT  = (long)NBH * HDM * SPAD;      // 2,752,512

  float* hbuf  = (float*)d_ws;
  float* tmp0  = hbuf + TD;
  float* big   = tmp0 + TD;    // qkv / padded scores / FFN activations
  float* qb    = big + TBIG;
  float* kb_   = qb + THD;
  float* vT    = kb_ + THD;
  float* ob    = vT + TVT;
  float* wsel  = ob + THD;                       // [E][T]
  float* projT = wsel + (long)NEXP * NTOK;       // 512*768
  int*   idx   = (int*)(projT + (long)NOUT * DIM);  // [E][T]
  int*   cnt   = idx + (long)NEXP * NTOK;        // [E]

  auto gemm = [&](const float* A, long lda, long sA,
                  const float* W, long ldw, long sW,
                  const float* bias, float* C, long ldc, long sC,
                  const float* res, long ldr, long sR,
                  const float* rs, const int* gA, const int* sC_, const int* dM,
                  int M, int N, int K, int act, int acc, int batches) {
    dim3 g((N + 63) / 64, (M + 127) / 128, batches);
    gemm_kernel<<<g, 256, 0, stream>>>(A, lda, sA, W, ldw, sW, bias,
                                       C, ldc, sC, res, ldr, sR, rs,
                                       gA, sC_, dM, M, N, K, act, acc);
  };

  auto attention = [&](const float* lw, const float* lb,
                       const float* iw, const float* ib,
                       const float* ow, const float* obias) {
    ln_kernel<<<NTOK, 256, 0, stream>>>(hbuf, DIM, lw, lb, tmp0, DIM, DIM);
    // qkv
    gemm(tmp0, DIM, 0, iw, DIM, 0, ib, big, 3 * DIM, 0,
         nullptr, 0, 0, nullptr, nullptr, nullptr, nullptr,
         NTOK, 3 * DIM, DIM, 0, 0, 1);
    {
      long tot = (long)NTOK * 3 * DIM;
      split_qkv_kernel<<<(int)((tot + 255) / 256), 256, 0, stream>>>(big, qb, kb_, vT);
    }
    // scores = q @ k^T  ([192] batches, 197x197 into 197x224-padded rows)
    gemm(qb, HDM, (long)SEQ * HDM, kb_, HDM, (long)SEQ * HDM, nullptr,
         big, SPAD, (long)SEQ * SPAD, nullptr, 0, 0,
         nullptr, nullptr, nullptr, nullptr, SEQ, SEQ, HDM, 0, 0, NBH);
    {
      long nrows = (long)NBH * SEQ;
      softmax_rows_kernel<<<(int)((nrows + 7) / 8), 256, 0, stream>>>(
          big, nrows, SEQ, SPAD, 0.125f);
    }
    // o = att @ v : A 197x224 (padded), B = vT [64 rows][224 k]
    gemm(big, SPAD, (long)SEQ * SPAD, vT, SPAD, (long)HDM * SPAD, nullptr,
         ob, HDM, (long)SEQ * HDM, nullptr, 0, 0,
         nullptr, nullptr, nullptr, nullptr, SEQ, HDM, SPAD, 0, 0, NBH);
    merge_heads_kernel<<<(int)((TD + 255) / 256), 256, 0, stream>>>(ob, tmp0);
    // h = h + o @ out_w^T + out_b
    gemm(tmp0, DIM, 0, ow, DIM, 0, obias, hbuf, DIM, 0,
         hbuf, DIM, 0, nullptr, nullptr, nullptr, nullptr,
         NTOK, DIM, DIM, 0, 0, 1);
  };

  // ---- one-time prep: v-pad zero, proj transpose, patch embed, cls/pos, ln_pre
  {
    long npad = (long)NBH * HDM * (SPAD - SEQ);
    zero_vpad_kernel<<<(int)((npad + 255) / 256), 256, 0, stream>>>(vT);
    long tp = (long)DIM * NOUT;
    transpose_proj_kernel<<<(int)((tp + 255) / 256), 256, 0, stream>>>(proj, projT);
    dim3 g(DIM / 64, PROWS / 64);
    patch_gemm_kernel<<<g, 256, 0, stream>>>(x, conv_w, hbuf);
    add_cls_pos_kernel<<<(int)((TD + 255) / 256), 256, 0, stream>>>(hbuf, cls_emb, pos_emb);
    ln_kernel<<<NTOK, 256, 0, stream>>>(hbuf, DIM, ln_pre_w, ln_pre_b, hbuf, DIM, DIM);
  }

  // ---- dense blocks ----
  for (int i = 0; i < 4; ++i) {
    attention(d_ln1_w + i * DIM, d_ln1_b + i * DIM,
              d_ain_w + (long)i * 3 * DIM * DIM, d_ain_b + (long)i * 3 * DIM,
              d_aout_w + (long)i * DIM * DIM, d_aout_b + (long)i * DIM);
    ln_kernel<<<NTOK, 256, 0, stream>>>(hbuf, DIM, d_ln2_w + i * DIM,
                                        d_ln2_b + i * DIM, tmp0, DIM, DIM);
    gemm(tmp0, DIM, 0, d_fc_w + (long)i * FFD * DIM, DIM, 0,
         d_fc_b + (long)i * FFD, big, FFD, 0, nullptr, 0, 0,
         nullptr, nullptr, nullptr, nullptr, NTOK, FFD, DIM, 1, 0, 1);
    gemm(big, FFD, 0, d_pj_w + (long)i * DIM * FFD, FFD, 0,
         d_pj_b + (long)i * DIM, hbuf, DIM, 0, hbuf, DIM, 0,
         nullptr, nullptr, nullptr, nullptr, NTOK, DIM, FFD, 0, 0, 1);
  }

  // ---- MoE blocks (top-2 gathered expert dispatch) ----
  for (int i = 0; i < 4; ++i) {
    attention(m_ln1_w + i * DIM, m_ln1_b + i * DIM,
              m_ain_w + (long)i * 3 * DIM * DIM, m_ain_b + (long)i * 3 * DIM,
              m_aout_w + (long)i * DIM * DIM, m_aout_b + (long)i * DIM);
    ln_kernel<<<NTOK, 256, 0, stream>>>(hbuf, DIM, m_ln2_w + i * DIM,
                                        m_ln2_b + i * DIM, tmp0, DIM, DIM);
    // router logits straight into d_out (feat occupies first BZ*NOUT floats)
    float* logits = out + (long)BZ * NOUT + (long)i * NTOK * NEXP;
    gemm(tmp0, DIM, 0, m_gate_w + (long)i * NEXP * DIM, DIM, 0, nullptr,
         logits, NEXP, 0, nullptr, 0, 0,
         nullptr, nullptr, nullptr, nullptr, NTOK, NEXP, DIM, 0, 0, 1);
    zero_cnt_kernel<<<1, NEXP, 0, stream>>>(cnt);
    route_topk_kernel<<<(NTOK + 255) / 256, 256, 0, stream>>>(logits, cnt, idx,
                                                              wsel, NTOK);
    for (int e = 0; e < NEXP; ++e) {
      long we_off = (long)(i * NEXP + e);
      const int*   idx_e = idx + (long)e * NTOK;
      const float* w_e   = wsel + (long)e * NTOK;
      const int*   cnt_e = cnt + e;
      // fc: gather selected token rows of tmp0, compact output rows in big
      gemm(tmp0, DIM, 0, m_fc_w + we_off * FFD * DIM, DIM, 0,
           m_fc_b + we_off * FFD, big, FFD, 0, nullptr, 0, 0,
           nullptr, idx_e, nullptr, cnt_e, NTOK, FFD, DIM, 1, 0, 1);
      // pj: compact rows in, scatter h[token] += w * (act @ pj_w^T + pj_b)
      gemm(big, FFD, 0, m_pj_w + we_off * DIM * FFD, FFD, 0,
           m_pj_b + we_off * DIM, hbuf, DIM, 0, nullptr, 0, 0,
           w_e, nullptr, idx_e, cnt_e, NTOK, DIM, FFD, 0, 1, 1);
    }
  }

  // ---- ln_post on CLS token + projection (via projT, standard path) ----
  ln_kernel<<<BZ, 256, 0, stream>>>(hbuf, (long)SEQ * DIM, ln_post_w, ln_post_b,
                                    tmp0, DIM, DIM);
  gemm(tmp0, DIM, 0, projT, DIM, 0, nullptr, out, NOUT, 0,
       nullptr, 0, 0, nullptr, nullptr, nullptr, nullptr,
       BZ, NOUT, DIM, 0, 0, 1);
}
